// Cost_map_transformer_13898514170478
// MI455X (gfx1250) — matrix operand order repaired and checked by reading, not compiled, not run on hardware
//
#include <hip/hip_runtime.h>
#include <hip/hip_bf16.h>

// ---- WMMA fragment types (gfx1250, wave32) ----
typedef __attribute__((ext_vector_type(16))) _Float16     v16h;
typedef __attribute__((ext_vector_type(8)))  float        v8f;
typedef __attribute__((ext_vector_type(4)))  unsigned int u32x4;

#define KD      41          // kernel size
#define PAD     20
#define IMG_H   512
#define IMG_W   512
#define NIMG    32
#define STRIPW  256         // output cols per block (16 tiles of 16)
#define IN_ROWS 56          // 16 + 40 halo rows
#define IN_COLS 304         // 16*16 + 48; 304*2B stride is 16B-aligned

// One block: 256 threads = 8 waves; covers a 16-row x 256-col output strip.
// Each wave owns two 16x16 output tiles (tx = wave and wave+8) with two
// live accumulators, so B fragments are loaded once per dy and the loop
// carries two independent WMMA chains.
__global__ __launch_bounds__(256)
void conv41_softplus_wmma(const float* __restrict__ in,
                          const float* __restrict__ wgt,
                          float* __restrict__ out) {
    // B fragments, pre-arranged in exact per-lane WMMA B layout:
    // [dy][chunk][lane][16 f16]  -> 41*2*32*16 f16 = 83,968 B
    __shared__ _Float16 sB[KD * 2 * 32 * 16];
    // Input halo strip in f16, zero-padded: 56 x 304 f16 = 34,048 B
    __shared__ _Float16 sI[IN_ROWS * IN_COLS];

    const int tid  = threadIdx.x;
    const int lane = tid & 31;
    const int wave = tid >> 5;

    const int bid   = blockIdx.x;
    const int half  = bid & 1;          // which 256-col half of the image
    const int strip = (bid >> 1) & 31;  // which 16-row strip
    const int img   = bid >> 6;         // image index

    const int y0 = strip * 16;          // first output row of this block
    const int xb = half * STRIPW;       // first output col of this block

    // ---------------- 1) Build B fragments (Toeplitz weight band) ----------------
    // For fragment (dy, c): B[k][n] = W[dy][c*32 + k - n], zero off-band.
    // Per-lane layout of 16x16x32 f16 B (32x16): lanes 0-15: N=lane, K = 0..7,16..23;
    // lanes 16-31: N=lane-16, K = 8..15,24..31 (matches A fragment order).
    for (int idx = tid; idx < KD * 2 * 32; idx += 256) {
        const int dy  = idx >> 6;
        const int rem = idx & 63;
        const int c   = rem >> 5;
        const int ln  = rem & 31;
        const int N   = ln & 15;
        const int kb  = c * 32 + ((ln & 16) ? 8 : 0);
        const float* wrow = wgt + dy * KD;
        _Float16* dst = &sB[idx * 16];
#pragma unroll
        for (int t = 0; t < 16; ++t) {
            const int wv = kb + t + ((t >= 8) ? 8 : 0) - N;
            const float v = (wv >= 0 && wv < KD) ? wrow[wv] : 0.0f;
            dst[t] = (_Float16)v;
        }
    }

    // ---------------- 2) Stage input halo (f32 -> f16) into LDS ----------------
    const float* inImg = in + (size_t)img * IMG_H * IMG_W;
    for (int q = tid; q < IN_ROWS * (IN_COLS / 4); q += 256) {
        const int r  = q / (IN_COLS / 4);
        const int c4 = (q % (IN_COLS / 4)) * 4;
        const int gy = y0 - PAD + r;
        _Float16 h4[4];
#pragma unroll
        for (int e = 0; e < 4; ++e) {
            const int gx = xb - PAD + c4 + e;
            float v = 0.0f;
            if (gy >= 0 && gy < IMG_H && gx >= 0 && gx < IMG_W)
                v = inImg[gy * IMG_W + gx];
            h4[e] = (_Float16)v;
        }
        // 8-byte aligned LDS store (IN_COLS and c4 are multiples of 4)
        *reinterpret_cast<unsigned long long*>(&sI[r * IN_COLS + c4]) =
            *reinterpret_cast<const unsigned long long*>(h4);
    }

    __syncthreads();

    // ---------------- 3) WMMA main loop (dy-outer, dual accumulators) ----------
    // A-fragment per-lane layout (16x32 f16, M x K):
    //   lanes 0-15 : M = lane,    elements 0..7 -> K = 0..7,  8..15 -> K = 16..23
    //   lanes 16-31: M = lane-16, elements 0..7 -> K = 8..15, 8..15 -> K = 24..31
    const int M   = lane & 15;
    const int akb = (lane & 16) ? 8 : 0;   // A K-run base within chunk
    const int Mo  = (lane & 16) ? 8 : 0;   // C/D layout M offset

    const int Lc0 = wave * 16;             // tile tx = wave
    const int Lc1 = Lc0 + 128;             // tile tx = wave + 8

    v8f acc0 = {0.f, 0.f, 0.f, 0.f, 0.f, 0.f, 0.f, 0.f};
    v8f acc1 = {0.f, 0.f, 0.f, 0.f, 0.f, 0.f, 0.f, 0.f};

    for (int dy = 0; dy < KD; ++dy) {
        const _Float16* arow = &sI[(M + dy) * IN_COLS];
        const _Float16* bfrag = &sB[((dy * 2) * 32 + lane) * 16];

        // B fragments for both K-chunks (shared by both tiles): 2x 32B LDS loads
        const v16h b0 = *reinterpret_cast<const v16h*>(bfrag);            // chunk 0
        const v16h b1 = *reinterpret_cast<const v16h*>(bfrag + 32 * 16);  // chunk 1

        // A fragments: 16B-aligned ds_load_b128 pairs
        v16h a00, a01, a10, a11;
        reinterpret_cast<u32x4*>(&a00)[0] = *reinterpret_cast<const u32x4*>(arow + Lc0 + akb);
        reinterpret_cast<u32x4*>(&a00)[1] = *reinterpret_cast<const u32x4*>(arow + Lc0 + akb + 16);
        reinterpret_cast<u32x4*>(&a01)[0] = *reinterpret_cast<const u32x4*>(arow + Lc0 + 32 + akb);
        reinterpret_cast<u32x4*>(&a01)[1] = *reinterpret_cast<const u32x4*>(arow + Lc0 + 32 + akb + 16);
        reinterpret_cast<u32x4*>(&a10)[0] = *reinterpret_cast<const u32x4*>(arow + Lc1 + akb);
        reinterpret_cast<u32x4*>(&a10)[1] = *reinterpret_cast<const u32x4*>(arow + Lc1 + akb + 16);
        reinterpret_cast<u32x4*>(&a11)[0] = *reinterpret_cast<const u32x4*>(arow + Lc1 + 32 + akb);
        reinterpret_cast<u32x4*>(&a11)[1] = *reinterpret_cast<const u32x4*>(arow + Lc1 + 32 + akb + 16);

        // Two independent accumulation chains, interleaved.
        acc0 = __builtin_amdgcn_wmma_f32_16x16x32_f16(false, a00, false, b0,
                                                      (short)0, acc0, false, false);
        acc1 = __builtin_amdgcn_wmma_f32_16x16x32_f16(false, a10, false, b0,
                                                      (short)0, acc1, false, false);
        acc0 = __builtin_amdgcn_wmma_f32_16x16x32_f16(false, a01, false, b1,
                                                      (short)0, acc0, false, false);
        acc1 = __builtin_amdgcn_wmma_f32_16x16x32_f16(false, a11, false, b1,
                                                      (short)0, acc1, false, false);
    }

    // ---------------- 4) softplus + store ----------------
    // C/D layout: VGPR v, lane l -> M = v + Mo, N = l & 15
    const int N = lane & 15;
#pragma unroll
    for (int v = 0; v < 8; ++v) {
        const float x0 = acc0[v];
        const float x1 = acc1[v];
        const float sp0 = fmaxf(x0, 0.0f) + log1pf(__expf(-fabsf(x0)));
        const float sp1 = fmaxf(x1, 0.0f) + log1pf(__expf(-fabsf(x1)));
        const size_t rowbase = ((size_t)img * IMG_H + (y0 + v + Mo)) * IMG_W;
        out[rowbase + (xb + Lc0 + N)] = sp0;
        out[rowbase + (xb + Lc1 + N)] = sp1;
    }
}

extern "C" void kernel_launch(void* const* d_in, const int* in_sizes, int n_in,
                              void* d_out, int out_size, void* d_ws, size_t ws_size,
                              hipStream_t stream) {
    const float* cost = (const float*)d_in[0];  // (32,1,512,512) f32
    // d_in[1] = TRANS_MAT, unused in conv mode (matches reference)
    const float* wgt  = (const float*)d_in[2];  // (1,1,41,41) f32
    float* out = (float*)d_out;                 // (32,1,512,512) f32

    const int nblocks = NIMG * 32 * 2;          // image * strip * half = 2048
    conv41_softplus_wmma<<<dim3(nblocks), dim3(256), 0, stream>>>(cost, wgt, out);
}
